// RNDModel_15762529976970
// MI455X (gfx1250) — compile-verified, matrix operand order checked by reading, not run on hardware
//
#include <hip/hip_runtime.h>
#include <hip/hip_bf16.h>
#include <stdint.h>

typedef __attribute__((ext_vector_type(16))) _Float16 v16h;
typedef __attribute__((ext_vector_type(8)))  _Float16 v8h;
typedef __attribute__((ext_vector_type(8)))  float    v8f;
typedef __attribute__((ext_vector_type(4)))  unsigned int v4u;
typedef __attribute__((ext_vector_type(8)))  int      v8i;
typedef __attribute__((ext_vector_type(4)))  int      v4i;

#define BATCH   256
#define NP      55
#define NDM     3
#define NH      128
#define HS      136                      // padded f16 row stride (halves) - kills bank conflicts
#define NL      5
#define EPB     (NP * (NP - 1))          // 2970 edges per batch graph
#define ETILES  ((EPB + 15) / 16)        // 186
#define NTHREADS 256
#define NWAVES   8
#define CRANGE   3.0f                    // 15.0 / L

// f16-prepacked (fragment-major swizzled) weight region sizes, in halves
#define SZ_EW1 (NL * 256 * NH)
#define SZ_EW2 (NL * NH * NH)
#define SZ_NW1 (NL * 256 * NH)
#define SZ_NW2 (NL * NH * NH)
#define SZ_CW1 (NL * NH * NH)
#define PER_NET_H (SZ_EW1 + SZ_EW2 + SZ_NW1 + SZ_NW2 + SZ_CW1)

// ---------------- LDS layout (byte offsets; dynamic LDS assumed base 0) ----
#define SOFF_W0   0u                         // 64KB slot: ew1 layer / nw1 layer
#define SOFF_W1   65536u                     // 32KB slot: ew2 layer / nw2 layer
#define SOFF_W2   98304u                     // 32KB slot: cw1 layer
#define OFF_HBUF  131072u                    // f32 [NP][NH]
#define OFF_AGG   (OFF_HBUF + NP * NH * 4u)  // 159232
#define OFF_H16   (OFF_AGG + NP * NH * 4u)   // 187392  f16 [NP][HS]
#define OFF_AGG16 (OFF_H16 + NP * HS * 2u)   // 202352  f16 [NP][HS]
#define OFF_XA    (OFF_AGG16 + NP * HS * 2u) // 217312  f32 [56][4]
#define OFF_XB    (OFF_XA + 56u * 4u * 4u)   // 218208
#define OFF_EATTR (OFF_XB + 56u * 4u * 4u)   // 219104  f32 [EPB]
#define OFF_LP    231000u                    // 9*NH f32 layer params (8B aligned: 230984->pad)
#define OFF_MEAN  (OFF_LP + 9u * NH * 4u + 8u)   // 235616
#define OFF_WSCR  (OFF_MEAN + 32u)               // 235648
#define WSCR_BYTES 8960u                     // buf1(4352)+buf2(4352)+rad/ea/att(192)+pad
#define SMEM_BYTES (OFF_WSCR + NWAVES * WSCR_BYTES)  // 307328

union AFrag { v16h v; v8h h8[2]; _Float16 h[16]; };
union CFrag { v8f  v; float f[8]; };

__device__ inline float sigmoidf_(float x) { return 1.f / (1.f + __expf(-x)); }
__device__ inline float siluf_(float x)    { return x * sigmoidf_(x); }

__device__ inline void wave_sync() {
  asm volatile("s_wait_dscnt 0" ::: "memory");
  __builtin_amdgcn_wave_barrier();
}

__device__ inline void tensor_wait() {
#if defined(__AMDGCN__) && __has_builtin(__builtin_amdgcn_s_wait_tensorcnt)
  __builtin_amdgcn_s_wait_tensorcnt(0);
#else
  asm volatile("s_wait_tensorcnt 0" ::: "memory");
#endif
}

#if defined(__AMDGCN__) && __has_builtin(__builtin_amdgcn_tensor_load_to_lds)
#define HAVE_TDM 1
// 2D TDM copy: rows of 4096 halves (one fragment-major "ktile block" per row).
__device__ inline void tdm_load_2d(unsigned lds_off, const _Float16* gsrc, unsigned halves) {
  const unsigned rows = halves >> 12;                  // halves / 4096
  const unsigned long long ga = (unsigned long long)(uintptr_t)gsrc;
  v4u g0 = { 1u,                                       // count=1
             lds_off,                                  // lds_addr
             (unsigned)(ga & 0xffffffffu),             // global_addr lo
             (unsigned)((ga >> 32) & 0x1ffffffu) | (2u << 30) };  // addr hi | type=2
  v8i g1 = { (int)(1u << 16),                          // data_size=1 (2B), mask=0
             (int)(4096u << 16),                       // tensor_dim0 lo16
             (int)(rows << 16),                        // dim0 hi=0 | tensor_dim1 lo16
             (int)(4096u << 16),                       // dim1 hi=0 | tile_dim0=4096
             (int)rows,                                // tile_dim1=rows, tile_dim2=0
             (int)4096,                                // tensor_dim0_stride lo32
             0, 0 };
  v4i z4 = {0, 0, 0, 0};
#if __clang_major__ >= 23
  v8i z8 = {0, 0, 0, 0, 0, 0, 0, 0};
  __builtin_amdgcn_tensor_load_to_lds(g0, g1, z4, z4, z8, 0);
#else
  __builtin_amdgcn_tensor_load_to_lds(g0, g1, z4, z4, 0);
#endif
}
#else
#define HAVE_TDM 0
#endif

__device__ inline void stage_copy(_Float16* dst, const _Float16* src, int halves, int tid) {
  for (int i = tid * 8; i < halves; i += NTHREADS * 8)
    *(v8h*)(dst + i) = *(const v8h*)(src + i);
}

// One-wave GEMM: 16xK A tile (per-lane 32B chunks supplied by `arow`) x Kx128
// f16 weights staged in LDS in fragment-major layout.  A fragments are
// register-resident and reused across all 8 N-tiles; every fragment load is a
// pair of ds_load_b128.  Layouts follow the CDNA5 wave32 WMMA VGPR tables.
template <int K, typename ARow, typename Epi>
__device__ inline void wave_gemm(const _Float16* __restrict__ wstage, ARow arow, Epi epi) {
  const int lane  = threadIdx.x & 31;
  const int ncol  = lane & 15;
  const int kg    = (lane >> 4) << 3;   // 0 or 8: K-half select
  const int mbase = (lane >> 4) << 3;   // C rows 0..7 or 8..15
  AFrag afr[K / 32];
#pragma unroll
  for (int kt = 0; kt < K; kt += 32) {
    const _Float16* ap = arow(kt);
    afr[kt >> 5].h8[0] = *(const v8h*)(ap + kg);
    afr[kt >> 5].h8[1] = *(const v8h*)(ap + kg + 16);
  }
  for (int nb = 0; nb < NH / 16; ++nb) {
    CFrag acc = {};
#pragma unroll
    for (int kt = 0; kt < K; kt += 32) {
      const _Float16* bp = wstage + (size_t)((((kt >> 5) << 3) + nb) * 32 + lane) * 16;
      AFrag bf;
      bf.h8[0] = *(const v8h*)(bp);
      bf.h8[1] = *(const v8h*)(bp + 8);
      acc.v = __builtin_amdgcn_wmma_f32_16x16x32_f16(
          false, afr[kt >> 5].v, false, bf.v, (short)0, acc.v, false, false);
    }
#pragma unroll
    for (int i = 0; i < 8; ++i) epi(mbase + i, nb * 16 + ncol, acc.f[i]);
  }
}

struct NetParams {
  const float *emb_w, *emb_b;
  const float *ew1, *eb1, *eb2, *aw, *ab, *nb1, *nb2, *cb1, *cw2;
  const _Float16 *ew1h, *ew2h, *nw1h, *nw2h, *cw1h;   // swizzled f16 in ws
};
struct KArgs {
  const float* x;
  NetParams net[2];
  float* feat;   // [2][BATCH][NP*NDM]
};

__global__ __launch_bounds__(NTHREADS) void egnn_kernel(KArgs args) {
  const int b    = blockIdx.x;
  const int neti = blockIdx.y;
  const NetParams& prm = args.net[neti];
  const int tid  = threadIdx.x;
  const int wid  = tid >> 5;
  const int lane = tid & 31;
  const int mrow = lane & 15;

  extern __shared__ char smem[];
  _Float16* stg_w0 = (_Float16*)(smem + SOFF_W0);
  _Float16* stg_w1 = (_Float16*)(smem + SOFF_W1);
  _Float16* stg_w2 = (_Float16*)(smem + SOFF_W2);
  float*    hbuf   = (float*)(smem + OFF_HBUF);
  float*    agg    = (float*)(smem + OFF_AGG);
  _Float16* h16    = (_Float16*)(smem + OFF_H16);
  _Float16* agg16  = (_Float16*)(smem + OFF_AGG16);
  float*    xA     = (float*)(smem + OFF_XA);
  float*    xB     = (float*)(smem + OFF_XB);
  float*    eattr  = (float*)(smem + OFF_EATTR);
  float*    lp     = (float*)(smem + OFF_LP);
  float*    meanv  = (float*)(smem + OFF_MEAN);
  char*     wb0    = smem + OFF_WSCR + (unsigned)wid * WSCR_BYTES;
  _Float16* buf1   = (_Float16*)wb0;                  // [16][HS]
  _Float16* buf2   = buf1 + 16 * HS;                  // [16][HS]
  float*    s_rad  = (float*)(buf2 + 16 * HS);
  float*    s_ea   = s_rad + 16;
  float*    s_att  = s_ea + 16;

  const float* xg = args.x + b * (NP * NDM);

  // ---- init coords + h0 = emb_w[0,:] + emb_b (one-hot type 0, t = 0) ----
  for (int i = tid; i < NP * NDM; i += NTHREADS) {
    const int p = i / 3, d = i % 3;
    xA[p * 4 + d] = xg[i];
  }
  for (int i = tid; i < NP * NH; i += NTHREADS) {
    const int p = i / NH, k = i % NH;
    const float h0 = prm.emb_w[k] + prm.emb_b[k];
    hbuf[i] = h0;
    h16[p * HS + k] = (_Float16)h0;
  }
  __syncthreads();
  for (int e = tid; e < EPB; e += NTHREADS) {
    const int r = e / 54, q = e % 54;
    const int c = q + (q >= r ? 1 : 0);
    const float dx = xA[r * 4 + 0] - xA[c * 4 + 0];
    const float dy = xA[r * 4 + 1] - xA[c * 4 + 1];
    const float dz = xA[r * 4 + 2] - xA[c * 4 + 2];
    eattr[e] = dx * dx + dy * dy + dz * dz;
  }
  __syncthreads();

  for (int l = 0; l < NL; ++l) {
    float* xr = (l & 1) ? xB : xA;
    float* xw = (l & 1) ? xA : xB;
    const _Float16* ew1h = prm.ew1h + (size_t)l * 256 * NH;
    const _Float16* ew2h = prm.ew2h + (size_t)l * NH * NH;
    const _Float16* cw1h = prm.cw1h + (size_t)l * NH * NH;
    const _Float16* nw1h = prm.nw1h + (size_t)l * 256 * NH;
    const _Float16* nw2h = prm.nw2h + (size_t)l * NH * NH;
    const float* tail0 = prm.ew1 + (size_t)(l * 258 + 256) * NH;  // radial row
    const float* tail1 = tail0 + NH;                              // edge_attr row
    const float  abl   = prm.ab[l];

    // ---- stage this layer's edge weights into LDS (TDM async DMA) ----
#if HAVE_TDM
    if (wid == 0) {
      tdm_load_2d(SOFF_W0, ew1h, 256 * NH);
      tdm_load_2d(SOFF_W1, ew2h, NH * NH);
      tdm_load_2d(SOFF_W2, cw1h, NH * NH);
    }
#else
    stage_copy(stg_w0, ew1h, 256 * NH, tid);
    stage_copy(stg_w1, ew2h, NH * NH, tid);
    stage_copy(stg_w2, cw1h, NH * NH, tid);
#endif
    // overlap: clear agg, copy coords, cache layer params in LDS
    for (int i = tid; i < NP * NH; i += NTHREADS) agg[i] = 0.f;
    for (int i = tid; i < NP * 4; i += NTHREADS) xw[i] = xr[i];
    for (int i = tid; i < NH; i += NTHREADS) {
      lp[0 * NH + i] = prm.eb1[l * NH + i];
      lp[1 * NH + i] = prm.eb2[l * NH + i];
      lp[2 * NH + i] = prm.aw [l * NH + i];
      lp[3 * NH + i] = prm.cb1[l * NH + i];
      lp[4 * NH + i] = prm.cw2[l * NH + i];
      lp[5 * NH + i] = prm.nb1[l * NH + i];
      lp[6 * NH + i] = prm.nb2[l * NH + i];
      lp[7 * NH + i] = tail0[i];
      lp[8 * NH + i] = tail1[i];
    }
#if HAVE_TDM
    if (wid == 0) tensor_wait();
#endif
    __syncthreads();

    // ---- edge tiles: 16 edges per wave-iteration ----
    for (int tile = wid; tile < ETILES; tile += NWAVES) {
      const int e0 = tile * 16;
      int em = e0 + mrow; if (em >= EPB) em = EPB - 1;   // clamp padded rows
      const int er = em / 54, eq = em % 54;
      const int ec = eq + (eq >= er ? 1 : 0);
      float dnx = 0.f, dny = 0.f, dnz = 0.f;
      if (lane < 16) {
        const float dx = xr[er * 4 + 0] - xr[ec * 4 + 0];
        const float dy = xr[er * 4 + 1] - xr[ec * 4 + 1];
        const float dz = xr[er * 4 + 2] - xr[ec * 4 + 2];
        const float rad = dx * dx + dy * dy + dz * dz;
        const float inv = 1.f / (sqrtf(rad) + 1.f);
        dnx = dx * inv; dny = dy * inv; dnz = dz * inv;
        s_rad[mrow] = rad;
        s_ea[mrow]  = eattr[em];
      }
      wave_sync();

      // m1 = silu([h_r || h_c] @ ew1[0:256] + radial*w256 + eattr*w257 + b)
      wave_gemm<256>(stg_w0,
        [&](int kt) -> const _Float16* {
          return (kt < NH) ? (h16 + er * HS + kt) : (h16 + ec * HS + (kt - NH));
        },
        [&](int m, int n, float v) {
          const float val = v + lp[0 * NH + n] + s_rad[m] * lp[7 * NH + n]
                              + s_ea[m] * lp[8 * NH + n];
          buf1[m * HS + n] = (_Float16)siluf_(val);
        });
      wave_sync();

      // m2 = silu(m1 @ ew2 + b)
      wave_gemm<128>(stg_w1,
        [&](int kt) -> const _Float16* { return buf1 + mrow * HS + kt; },
        [&](int m, int n, float v) {
          buf2[m * HS + n] = (_Float16)siluf_(v + lp[1 * NH + n]);
        });
      wave_sync();

      // attention gate; fold into buf2 in place
      if (lane < 16) {
        float s = abl;
        for (int k = 0; k < NH; ++k) s += (float)buf2[mrow * HS + k] * lp[2 * NH + k];
        s_att[mrow] = sigmoidf_(s);
      }
      wave_sync();
      for (int m = 0; m < 16; ++m) {
        const float am = s_att[m];
        for (int n = lane; n < NH; n += 32)
          buf2[m * HS + n] = (_Float16)((float)buf2[m * HS + n] * am);
      }
      wave_sync();

      // segment_sum(m, rows) into agg (ds_add_f32 atomics)
      for (int m = 0; m < 16; ++m) {
        const int e = e0 + m;
        if (e >= EPB) break;
        const int r = e / 54;
        for (int qq = 0; qq < 4; ++qq) {
          const int n = lane + qq * 32;
          atomicAdd(&agg[r * NH + n], (float)buf2[m * HS + n]);
        }
      }

      // cmid = silu(m @ cw1 + b)
      wave_gemm<128>(stg_w2,
        [&](int kt) -> const _Float16* { return buf2 + mrow * HS + kt; },
        [&](int m, int n, float v) {
          buf1[m * HS + n] = (_Float16)siluf_(v + lp[3 * NH + n]);
        });
      wave_sync();

      // s = tanh(cmid @ cw2) * range; equivariant coord scatter
      if (lane < 16 && (e0 + mrow) < EPB) {
        float s = 0.f;
        for (int k = 0; k < NH; ++k) s += (float)buf1[mrow * HS + k] * lp[4 * NH + k];
        s = tanhf(s) * CRANGE;
        atomicAdd(&xw[er * 4 + 0], dnx * s);
        atomicAdd(&xw[er * 4 + 1], dny * s);
        atomicAdd(&xw[er * 4 + 2], dnz * s);
      }
    }
    __syncthreads();

    // ---- stage node weights (reuse edge slots), convert agg -> f16 ----
#if HAVE_TDM
    if (wid == 0) {
      tdm_load_2d(SOFF_W0, nw1h, 256 * NH);
      tdm_load_2d(SOFF_W1, nw2h, NH * NH);
    }
#else
    stage_copy(stg_w0, nw1h, 256 * NH, tid);
    stage_copy(stg_w1, nw2h, NH * NH, tid);
#endif
    for (int i = tid; i < NP * NH; i += NTHREADS)
      agg16[(i / NH) * HS + (i % NH)] = (_Float16)agg[i];
#if HAVE_TDM
    if (wid == 0) tensor_wait();
#endif
    __syncthreads();

    // ---- node MLP: 4 tiles of 16 nodes on waves 0..3; residual into h ----
    if (wid < 4) {
      int node = wid * 16 + mrow; if (node >= NP) node = NP - 1;
      wave_gemm<256>(stg_w0,
        [&](int kt) -> const _Float16* {
          return (kt < NH) ? (h16 + node * HS + kt) : (agg16 + node * HS + (kt - NH));
        },
        [&](int m, int n, float v) {
          buf1[m * HS + n] = (_Float16)siluf_(v + lp[5 * NH + n]);
        });
      wave_sync();
      wave_gemm<128>(stg_w1,
        [&](int kt) -> const _Float16* { return buf1 + mrow * HS + kt; },
        [&](int m, int n, float v) {
          const int nd = wid * 16 + m;
          if (nd < NP) {
            const float hn = hbuf[nd * NH + n] + v + lp[6 * NH + n];
            hbuf[nd * NH + n] = hn;
            h16[nd * HS + n] = (_Float16)hn;
          }
        });
    }
    __syncthreads();
  }

  // ---- vel = x_final - x0, remove per-batch center-of-mass drift ----
  float* xf = xB;  // NL = 5 (odd): final coords in xB
  if (tid < 3) {
    float s = 0.f;
    for (int p = 0; p < NP; ++p) s += xf[p * 4 + tid] - xg[p * 3 + tid];
    meanv[tid] = s / (float)NP;
  }
  __syncthreads();
  float* featp = args.feat + (size_t)(neti * BATCH + b) * (NP * NDM);
  for (int i = tid; i < NP * NDM; i += NTHREADS) {
    const int p = i / 3, d = i % 3;
    featp[i] = (xf[p * 4 + d] - xg[i]) - meanv[d];
  }
}

// ---- prepack: f32 -> f16, fragment-major swizzle so a lane's B-fragment is
// 32 contiguous bytes: dst[((ktf*8 + nb)*32 + lane)*16 + j] = w[kk][n]. ----
__device__ inline void swz_decode(int t, int& kk, int& n) {
  const int j    = t & 15;
  const int lane = (t >> 4) & 31;
  const int nb   = (t >> 9) & 7;
  const int ktf  = t >> 12;
  kk = ktf * 32 + ((lane >> 4) << 3) + (j < 8 ? j : j + 8);
  n  = nb * 16 + (lane & 15);
}

struct PackArgs {
  const float* ew1[2];
  const float* ew2[2];
  const float* nw1[2];
  const float* nw2[2];
  const float* cw1[2];
  _Float16* dst;
};

__global__ void prepack_kernel(PackArgs pa) {
  const long total = 2L * PER_NET_H;
  for (long idx = (long)blockIdx.x * blockDim.x + threadIdx.x; idx < total;
       idx += (long)gridDim.x * blockDim.x) {
    const int netid = (int)(idx / PER_NET_H);
    int r = (int)(idx % PER_NET_H);
    int kk, n;
    float v;
    if (r < SZ_EW1) {
      const int l = r / (256 * NH);
      swz_decode(r % (256 * NH), kk, n);
      v = pa.ew1[netid][(size_t)(l * 258 + kk) * NH + n];
    } else if ((r -= SZ_EW1) < SZ_EW2) {
      const int l = r / (NH * NH);
      swz_decode(r % (NH * NH), kk, n);
      v = pa.ew2[netid][(size_t)(l * NH + kk) * NH + n];
    } else if ((r -= SZ_EW2) < SZ_NW1) {
      const int l = r / (256 * NH);
      swz_decode(r % (256 * NH), kk, n);
      v = pa.nw1[netid][(size_t)(l * 256 + kk) * NH + n];
    } else if ((r -= SZ_NW1) < SZ_NW2) {
      const int l = r / (NH * NH);
      swz_decode(r % (NH * NH), kk, n);
      v = pa.nw2[netid][(size_t)(l * NH + kk) * NH + n];
    } else {
      r -= SZ_NW2;
      const int l = r / (NH * NH);
      swz_decode(r % (NH * NH), kk, n);
      v = pa.cw1[netid][(size_t)(l * NH + kk) * NH + n];
    }
    pa.dst[idx] = (_Float16)v;
  }
}

__global__ void rnd_loss_kernel(const float* __restrict__ feat, float* __restrict__ out) {
  const int b = blockIdx.x * blockDim.x + threadIdx.x;
  if (b >= BATCH) return;
  const float* ft = feat + (size_t)b * (NP * NDM);                 // net 0 = params_t
  const float* fp = feat + (size_t)(BATCH + b) * (NP * NDM);       // net 1 = params_p
  float s = 0.f;
  for (int i = 0; i < NP * NDM; ++i) {
    const float d = fp[i] - ft[i];
    s += d * d;
  }
  out[b] = s;
}

extern "C" void kernel_launch(void* const* d_in, const int* in_sizes, int n_in,
                              void* d_out, int out_size, void* d_ws, size_t ws_size,
                              hipStream_t stream) {
  (void)in_sizes; (void)n_in; (void)out_size; (void)ws_size;
  // inputs: 0:x 1:rows 2:cols, params_t leaves [3..17], params_p [18..32]
  // leaf order: emb_w emb_b ew1 eb1 ew2 eb2 aw ab nw1 nb1 nw2 nb2 cw1 cb1 cw2
  const float* x = (const float*)d_in[0];
  _Float16* wsh = (_Float16*)d_ws;
  float* feat = (float*)((char*)d_ws + (size_t)2 * PER_NET_H * sizeof(_Float16));

  PackArgs pa;
  KArgs ka;
  ka.x = x;
  ka.feat = feat;
  const int bases[2] = {3, 18};
  for (int n = 0; n < 2; ++n) {
    const int bs = bases[n];
    auto leaf = [&](int i) { return (const float*)d_in[bs + i]; };
    pa.ew1[n] = leaf(2);  pa.ew2[n] = leaf(4);  pa.nw1[n] = leaf(8);
    pa.nw2[n] = leaf(10); pa.cw1[n] = leaf(12);
    NetParams& np_ = ka.net[n];
    np_.emb_w = leaf(0);  np_.emb_b = leaf(1);
    np_.ew1 = leaf(2);    np_.eb1 = leaf(3);
    np_.eb2 = leaf(5);    np_.aw = leaf(6);  np_.ab = leaf(7);
    np_.nb1 = leaf(9);    np_.nb2 = leaf(11);
    np_.cb1 = leaf(13);   np_.cw2 = leaf(14);
    _Float16* nb = wsh + (size_t)n * PER_NET_H;
    np_.ew1h = nb;
    np_.ew2h = nb + SZ_EW1;
    np_.nw1h = nb + SZ_EW1 + SZ_EW2;
    np_.nw2h = nb + SZ_EW1 + SZ_EW2 + SZ_NW1;
    np_.cw1h = nb + SZ_EW1 + SZ_EW2 + SZ_NW1 + SZ_NW2;
  }
  pa.dst = wsh;

  prepack_kernel<<<512, 256, 0, stream>>>(pa);

  (void)hipFuncSetAttribute((const void*)egnn_kernel,
                            hipFuncAttributeMaxDynamicSharedMemorySize,
                            (int)SMEM_BYTES);
  egnn_kernel<<<dim3(BATCH, 2), NTHREADS, SMEM_BYTES, stream>>>(ka);

  rnd_loss_kernel<<<1, BATCH, 0, stream>>>(feat, (float*)d_out);
}
